// RationalBasicBlock_26654567039145
// MI455X (gfx1250) — matrix operand order repaired and checked by reading, not compiled, not run on hardware
//
#include <hip/hip_runtime.h>
#include <stdint.h>

// Problem constants
#define CIN   256
#define COUT  256
#define HH    56
#define WWID  56
#define HWX   3136      // 56*56
#define NSP   100352    // 32*3136 spatial columns
#define KTOT  2304      // 256*9
#define BK    32        // K step (wmma bf16 K)
#define BMT   128       // Cout tile per block
#define BNT   256       // spatial tile per block
#define LDSROW 40       // padded LDS row in halves (32 data + 8 pad) = 80B

typedef __attribute__((ext_vector_type(8)))  __bf16 vbf8;
typedef __attribute__((ext_vector_type(16))) __bf16 vbf16;
typedef __attribute__((ext_vector_type(8)))  float  vf8;

__device__ __forceinline__ uint16_t f2bf(float f) {
    unsigned u = __float_as_uint(f);
    unsigned r = u + 0x7FFFu + ((u >> 16) & 1u);   // round-to-nearest-even
    return (uint16_t)(r >> 16);
}

__device__ __forceinline__ float rational_eval(float x, const float* nc, const float* dc) {
    float num = nc[5];
#pragma unroll
    for (int i = 4; i >= 0; --i) num = fmaf(num, x, nc[i]);
    float den = dc[3];
#pragma unroll
    for (int j = 2; j >= 0; --j) den = fmaf(den, x, dc[j]);
    float q = 1.0f + fabsf(x * den);
    return num / q;
}

// ---------------- zero scratch ----------------
__global__ void zero_f32_k(float* p, int n) {
    int i = blockIdx.x * 256 + threadIdx.x;
    if (i < n) p[i] = 0.0f;
}

// ---------------- pack weights: [Cout][Cin][3][3] f32 -> [Cout][rc*256+ci] bf16 ----------------
__global__ __launch_bounds__(256) void pack_w_k(const float* __restrict__ w1,
                                                const float* __restrict__ w2,
                                                uint16_t* __restrict__ A1,
                                                uint16_t* __restrict__ A2) {
    int gid = blockIdx.x * 256 + threadIdx.x;         // 0 .. 589823
    const float* w = blockIdx.y ? w2 : w1;
    uint16_t* A = blockIdx.y ? A2 : A1;
    int co = gid / KTOT;
    int t  = gid - co * KTOT;
    int rc = t >> 8;          // kh*3+kw
    int ci = t & 255;
    float v = w[(size_t)co * KTOT + (size_t)ci * 9 + rc];
    A[(size_t)co * KTOT + t] = f2bf(v);
}

// ---------------- x NCHW f32 -> NHWC bf16 (LDS tile transpose) ----------------
__global__ __launch_bounds__(256) void pack_x_k(const float* __restrict__ X,
                                                uint16_t* __restrict__ Xn) {
    __shared__ float tile[64 * 65];
    const int n   = blockIdx.z;
    const int co0 = blockIdx.y * 64;
    const int hw0 = blockIdx.x * 64;
    const int t   = threadIdx.x;
    const int cl  = t & 63;
    const int rb  = t >> 6;
    const float* src = X + (size_t)n * CIN * HWX;
#pragma unroll
    for (int i = 0; i < 16; i++) {
        int r = rb + i * 4;
        tile[r * 65 + cl] = src[(size_t)(co0 + r) * HWX + hw0 + cl];
    }
    __syncthreads();
#pragma unroll
    for (int i = 0; i < 16; i++) {
        int sl = rb + i * 4;
        Xn[((size_t)n * HWX + hw0 + sl) * CIN + co0 + cl] = f2bf(tile[cl * 65 + sl]);
    }
}

// ---------------- implicit-GEMM conv3x3 via WMMA bf16 + async-to-LDS staging ----------------
// Xn: [NSP][CIN] bf16 (NHWC), A: [COUT][KTOT] bf16, Y: [COUT][NSP] f32
__global__ __launch_bounds__(256) void conv3x3_wmma_k(const __bf16* __restrict__ Xn,
                                                      const __bf16* __restrict__ A,
                                                      float* __restrict__ Y) {
    __shared__ __bf16 As[2][BMT * LDSROW];   // 2 x 10240 B
    __shared__ __bf16 Bs[2][BNT * LDSROW];   // 2 x 20480 B

    const int tid = threadIdx.x;
    const int s0  = blockIdx.x * BNT;
    const int co0 = blockIdx.y * BMT;

    // loader mapping: each thread owns 16B chunk 'sub' of 4 B rows / 2 A rows
    const int sub = tid & 3;
    const int r0  = tid >> 2;            // 0..63

    long bbase[4];
    int  hwpack[4];                      // h | (w<<16)
#pragma unroll
    for (int j = 0; j < 4; j++) {
        int s  = s0 + r0 + 64 * j;
        int n  = s / HWX;
        int hw = s - n * HWX;
        int h  = hw / WWID;
        int w  = hw - h * WWID;
        hwpack[j] = h | (w << 16);
        bbase[j]  = ((long)n * HWX + (long)h * WWID + w) * CIN + sub * 8;
    }
    const __bf16* abase = A + (size_t)(co0 + r0) * KTOT + sub * 8;

    const int lane = tid & 31;
    const int wvid = tid >> 5;
    const int wm   = wvid & 1;      // 0..1 -> +64 rows
    const int wn   = wvid >> 1;     // 0..3 -> +64 cols
    const int half = lane >> 4;
    const int l16  = lane & 15;

    vf8 acc[4][4];
#pragma unroll
    for (int a = 0; a < 4; a++)
#pragma unroll
        for (int b = 0; b < 4; b++)
#pragma unroll
            for (int q = 0; q < 8; q++) acc[a][b][q] = 0.0f;

    auto load_tile = [&](int buf, int kc) {
        const int k0  = kc * BK;
        const int rc  = kc >> 3;                         // (kh,kw) block
        const int rr  = rc / 3 - 1;
        const int cc  = rc - (rc / 3) * 3 - 1;
        const int ci0 = (kc & 7) * BK;
        const int delta = (rr * WWID + cc) * CIN + ci0;  // wave-uniform im2col shift
        // A tile: rows r0, r0+64 — always valid, async copy straight to LDS
#pragma unroll
        for (int i = 0; i < 2; i++) {
            unsigned loff = (unsigned)(size_t)&As[buf][(r0 + 64 * i) * LDSROW + sub * 8];
            const __bf16* gp = abase + (size_t)i * 64 * KTOT + k0;
            asm volatile("global_load_async_to_lds_b128 %0, %1, off"
                         :: "v"(loff), "v"(gp) : "memory");
        }
        // B tile: rows r0 + 64*j — async when inside image, zero-fill LDS on halo
#pragma unroll
        for (int j = 0; j < 4; j++) {
            int h = hwpack[j] & 0xFFFF;
            int w = hwpack[j] >> 16;
            unsigned loff = (unsigned)(size_t)&Bs[buf][(r0 + 64 * j) * LDSROW + sub * 8];
            if ((unsigned)(h + rr) < HH && (unsigned)(w + cc) < WWID) {
                const __bf16* gp = Xn + bbase[j] + delta;
                asm volatile("global_load_async_to_lds_b128 %0, %1, off"
                             :: "v"(loff), "v"(gp) : "memory");
            } else {
                *(uint4*)(&Bs[buf][(r0 + 64 * j) * LDSROW + sub * 8]) = make_uint4(0u, 0u, 0u, 0u);
            }
        }
    };

    auto compute_tile = [&](int buf) {
        vbf16 af[4];
#pragma unroll
        for (int mt = 0; mt < 4; mt++) {
            const __bf16* rp = &As[buf][(wm * 64 + mt * 16 + l16) * LDSROW];
            vbf8 lo = *(const vbf8*)(rp + half * 8);
            vbf8 hi = *(const vbf8*)(rp + 16 + half * 8);
            af[mt] = __builtin_shufflevector(lo, hi, 0, 1, 2, 3, 4, 5, 6, 7,
                                             8, 9, 10, 11, 12, 13, 14, 15);
        }
#pragma unroll
        for (int nt = 0; nt < 4; nt++) {
            const __bf16* cp = &Bs[buf][(wn * 64 + nt * 16 + l16) * LDSROW + half * 8];
            vbf8 lo = *(const vbf8*)(cp);
            vbf8 hi = *(const vbf8*)(cp + 16);
            vbf16 bfr = __builtin_shufflevector(lo, hi, 0, 1, 2, 3, 4, 5, 6, 7,
                                                8, 9, 10, 11, 12, 13, 14, 15);
#pragma unroll
            for (int mt = 0; mt < 4; mt++)
                acc[mt][nt] = __builtin_amdgcn_wmma_f32_16x16x32_bf16(
                    false, af[mt], false, bfr, (short)0, acc[mt][nt], false, false);
        }
    };

    load_tile(0, 0);
    asm volatile("s_wait_asynccnt 0x0" ::: "memory");
    __syncthreads();
    int buf = 0;
    for (int kc = 0; kc < KTOT / BK; ++kc) {
        if (kc + 1 < KTOT / BK) load_tile(buf ^ 1, kc + 1);
        compute_tile(buf);
        asm volatile("s_wait_asynccnt 0x0" ::: "memory");
        __syncthreads();
        buf ^= 1;
    }

    // epilogue: C/D layout — lane half selects M+8, l16 = column
#pragma unroll
    for (int mt = 0; mt < 4; mt++) {
#pragma unroll
        for (int nt = 0; nt < 4; nt++) {
            int co = co0 + wm * 64 + mt * 16 + half * 8;
            int sc = s0 + wn * 64 + nt * 16 + l16;
            float* yp = Y + (size_t)co * NSP + sc;
#pragma unroll
            for (int v = 0; v < 8; v++) yp[(size_t)v * NSP] = acc[mt][nt][v];
        }
    }
}

// ---------------- per-channel sum / sumsq for training-mode BN ----------------
__global__ __launch_bounds__(256) void col_reduce_k(const float* __restrict__ Y,
                                                    float* __restrict__ sum,
                                                    float* __restrict__ sq) {
    const int co  = blockIdx.x;
    const int tid = threadIdx.x;
    const float* row = Y + (size_t)co * NSP + (size_t)blockIdx.y * 2048;
    float s = 0.0f, s2 = 0.0f;
#pragma unroll
    for (int i = 0; i < 8; i++) {
        float v = row[i * 256 + tid];
        s += v;
        s2 = fmaf(v, v, s2);
    }
    for (int off = 16; off > 0; off >>= 1) {
        s  += __shfl_xor(s, off, 32);
        s2 += __shfl_xor(s2, off, 32);
    }
    __shared__ float rs[8], rs2[8];
    int wid = tid >> 5;
    if ((tid & 31) == 0) { rs[wid] = s; rs2[wid] = s2; }
    __syncthreads();
    if (tid == 0) {
        float t = 0.0f, t2 = 0.0f;
#pragma unroll
        for (int i = 0; i < 8; i++) { t += rs[i]; t2 += rs2[i]; }
        atomicAdd(&sum[co], t);
        atomicAdd(&sq[co], t2);
    }
}

// ---------------- BN + rational, fused with CS->NHWC bf16 transpose ----------------
__global__ __launch_bounds__(256) void bn_act_pack_k(const float* __restrict__ Y,
                                                     const float* __restrict__ sum,
                                                     const float* __restrict__ sq,
                                                     const float* __restrict__ gamma,
                                                     const float* __restrict__ beta,
                                                     const float* __restrict__ ncoef,
                                                     const float* __restrict__ dcoef,
                                                     uint16_t* __restrict__ Out) {
    __shared__ float tile[64 * 65];
    const int co0 = blockIdx.y * 64;
    const int s0  = blockIdx.x * 64;
    const int t   = threadIdx.x;
    const int cl  = t & 63;
    const int rb  = t >> 6;
#pragma unroll
    for (int i = 0; i < 16; i++) {
        int r = rb + i * 4;
        tile[r * 65 + cl] = Y[(size_t)(co0 + r) * NSP + s0 + cl];
    }
    __syncthreads();
    const int co    = co0 + cl;                 // fixed per thread in write phase
    const float inv = 1.0f / (float)NSP;
    float mean = sum[co] * inv;
    float var  = sq[co] * inv - mean * mean;
    float sc   = gamma[co] * rsqrtf(var + 1e-5f);
    float sh   = fmaf(-mean, sc, beta[co]);
    const int g = co >> 6;
    float nc[6], dc[4];
#pragma unroll
    for (int i = 0; i < 6; i++) nc[i] = ncoef[g * 6 + i];
#pragma unroll
    for (int i = 0; i < 4; i++) dc[i] = dcoef[g * 4 + i];
#pragma unroll
    for (int i = 0; i < 16; i++) {
        int sl = rb + i * 4;
        float v = fmaf(tile[cl * 65 + sl], sc, sh);
        v = rational_eval(v, nc, dc);
        Out[(size_t)(s0 + sl) * CIN + co] = f2bf(v);
    }
}

// ---------------- BN2 + identity shortcut + rational -> NCHW f32 output ----------------
__global__ __launch_bounds__(256) void bn_add_act_out_k(const float* __restrict__ Y,
                                                        const float* __restrict__ sum,
                                                        const float* __restrict__ sq,
                                                        const float* __restrict__ gamma,
                                                        const float* __restrict__ beta,
                                                        const float* __restrict__ ncoef,
                                                        const float* __restrict__ dcoef,
                                                        const float* __restrict__ X,
                                                        float* __restrict__ Out) {
    const int gid = blockIdx.x * 256 + threadIdx.x;     // one float4 each
    const int co  = gid / (NSP / 4);
    const int rem = gid - co * (NSP / 4);
    const int s   = rem * 4;
    const int n   = s / HWX;
    const int hw  = s - n * HWX;
    const float4 y4 = *(const float4*)(Y + (size_t)co * NSP + s);
    const size_t xoff = (size_t)n * CIN * HWX + (size_t)co * HWX + hw;
    const float4 x4 = *(const float4*)(X + xoff);
    const float inv = 1.0f / (float)NSP;
    float mean = sum[co] * inv;
    float var  = sq[co] * inv - mean * mean;
    float sc   = gamma[co] * rsqrtf(var + 1e-5f);
    float sh   = fmaf(-mean, sc, beta[co]);
    const int g = co >> 6;
    float nc[6], dc[4];
#pragma unroll
    for (int i = 0; i < 6; i++) nc[i] = ncoef[g * 6 + i];
#pragma unroll
    for (int i = 0; i < 4; i++) dc[i] = dcoef[g * 4 + i];
    float4 o;
    o.x = rational_eval(fmaf(y4.x, sc, sh) + x4.x, nc, dc);
    o.y = rational_eval(fmaf(y4.y, sc, sh) + x4.y, nc, dc);
    o.z = rational_eval(fmaf(y4.z, sc, sh) + x4.z, nc, dc);
    o.w = rational_eval(fmaf(y4.w, sc, sh) + x4.w, nc, dc);
    *(float4*)(Out + xoff) = o;
}

extern "C" void kernel_launch(void* const* d_in, const int* in_sizes, int n_in,
                              void* d_out, int out_size, void* d_ws, size_t ws_size,
                              hipStream_t stream) {
    (void)in_sizes; (void)n_in; (void)out_size; (void)ws_size;
    const float* x   = (const float*)d_in[0];
    const float* w1  = (const float*)d_in[1];
    const float* g1  = (const float*)d_in[2];
    const float* be1 = (const float*)d_in[3];
    const float* w2  = (const float*)d_in[4];
    const float* g2  = (const float*)d_in[5];
    const float* be2 = (const float*)d_in[6];
    const float* ncf = (const float*)d_in[7];
    const float* dcf = (const float*)d_in[8];
    float* out = (float*)d_out;

    char* ws = (char*)d_ws;
    size_t off = 0;
    auto walloc = [&](size_t bytes) -> void* {
        void* p = ws + off;
        off += (bytes + 255) & ~(size_t)255;
        return p;
    };
    uint16_t* A1 = (uint16_t*)walloc((size_t)COUT * KTOT * 2);
    uint16_t* A2 = (uint16_t*)walloc((size_t)COUT * KTOT * 2);
    uint16_t* Xn = (uint16_t*)walloc((size_t)NSP * CIN * 2);      // NHWC bf16 (reused)
    float*    Y  = (float*)walloc((size_t)COUT * NSP * 4);        // CS f32 (reused)
    float* sums  = (float*)walloc(4 * 256 * 4);
    float *sum1 = sums, *sq1 = sums + 256, *sum2 = sums + 512, *sq2 = sums + 768;

    zero_f32_k<<<4, 256, 0, stream>>>(sums, 1024);
    pack_w_k<<<dim3(2304, 2), 256, 0, stream>>>(w1, w2, A1, A2);
    pack_x_k<<<dim3(49, 4, 32), 256, 0, stream>>>(x, Xn);

    conv3x3_wmma_k<<<dim3(NSP / BNT, COUT / BMT), 256, 0, stream>>>(
        (const __bf16*)Xn, (const __bf16*)A1, Y);
    col_reduce_k<<<dim3(256, 49), 256, 0, stream>>>(Y, sum1, sq1);
    bn_act_pack_k<<<dim3(NSP / 64, 4), 256, 0, stream>>>(Y, sum1, sq1, g1, be1, ncf, dcf, Xn);

    conv3x3_wmma_k<<<dim3(NSP / BNT, COUT / BMT), 256, 0, stream>>>(
        (const __bf16*)Xn, (const __bf16*)A2, Y);
    col_reduce_k<<<dim3(256, 49), 256, 0, stream>>>(Y, sum2, sq2);
    bn_add_act_out_k<<<(NSP / 4) * COUT / 256, 256, 0, stream>>>(
        Y, sum2, sq2, g2, be2, ncf, dcf, x, out);
}
